// MEMOL_48052094107931
// MI455X (gfx1250) — matrix operand order repaired and checked by reading, not compiled, not run on hardware
//
#include <hip/hip_runtime.h>
#include <hip/hip_bf16.h>
#include <math.h>

typedef __attribute__((ext_vector_type(16))) _Float16 v16h;
typedef __attribute__((ext_vector_type(8)))  _Float16 v8h;
typedef __attribute__((ext_vector_type(4)))  _Float16 v4h;
typedef __attribute__((ext_vector_type(8)))  float    v8f;

#define TOK   65536
#define DIMC  256
#define NEXP  4
#define NHEAD 4
#define DHEAD 64
#define HIDC  1024

__device__ __forceinline__ float wave_sum(float v) {
    v += __shfl_xor(v, 16, 32);
    v += __shfl_xor(v, 8, 32);
    v += __shfl_xor(v, 4, 32);
    v += __shfl_xor(v, 2, 32);
    v += __shfl_xor(v, 1, 32);
    return v;
}

// ---------------------------------------------------------------- weight cvt
__global__ void cvt_f32_f16_kernel(const float* __restrict__ in,
                                   _Float16* __restrict__ out, int n) {
    int i = blockIdx.x * blockDim.x + threadIdx.x;
    if (i < n) out[i] = (_Float16)in[i];
}

// ------------------------------------------------- LN1 + gate softmax + top2
// Writes xs[tok, e*256 + k] = comb[tok,e] * layernorm1(x)[tok,k]  (f16)
__global__ __launch_bounds__(256)
void ln_gate_kernel(const float* __restrict__ x,
                    const float* __restrict__ g1, const float* __restrict__ b1,
                    const float* __restrict__ gw, const float* __restrict__ gb,
                    _Float16* __restrict__ xs) {
    const int lane = threadIdx.x & 31;
    const int tok  = blockIdx.x * 8 + (threadIdx.x >> 5);
    const int kb   = lane * 8;
    const float* xr = x + (size_t)tok * DIMC;

    float v[8];
#pragma unroll
    for (int i = 0; i < 8; i++) v[i] = xr[kb + i];
    float s = 0.f;
#pragma unroll
    for (int i = 0; i < 8; i++) s += v[i];
    const float mean = wave_sum(s) * (1.f / 256.f);
    float q = 0.f;
#pragma unroll
    for (int i = 0; i < 8; i++) { float d = v[i] - mean; q += d * d; }
    const float rstd = rsqrtf(wave_sum(q) * (1.f / 256.f) + 1e-5f);

    float xn[8];
#pragma unroll
    for (int i = 0; i < 8; i++)
        xn[i] = (v[i] - mean) * rstd * g1[kb + i] + b1[kb + i];

    float lg[NEXP];
#pragma unroll
    for (int e = 0; e < NEXP; e++) {
        float p = 0.f;
#pragma unroll
        for (int i = 0; i < 8; i++) p += xn[i] * gw[(kb + i) * NEXP + e];
        lg[e] = wave_sum(p) + gb[e];
    }
    float mx = fmaxf(fmaxf(lg[0], lg[1]), fmaxf(lg[2], lg[3]));
    float pe[NEXP], ps = 0.f;
#pragma unroll
    for (int e = 0; e < NEXP; e++) { pe[e] = __expf(lg[e] - mx); ps += pe[e]; }
    const float inv = 1.f / ps;
#pragma unroll
    for (int e = 0; e < NEXP; e++) pe[e] *= inv;
    int i0 = 0;
#pragma unroll
    for (int e = 1; e < NEXP; e++) if (pe[e] > pe[i0]) i0 = e;
    int i1 = (i0 == 0) ? 1 : 0;
#pragma unroll
    for (int e = 0; e < NEXP; e++)
        if (e != i0 && e != i1 && pe[e] > pe[i1]) i1 = e;

    _Float16* xo = xs + (size_t)tok * (NEXP * DIMC);
#pragma unroll
    for (int e = 0; e < NEXP; e++) {
        const float c = (e == i0 || e == i1) ? pe[e] : 0.f;
        v8h h;
#pragma unroll
        for (int i = 0; i < 8; i++) h[i] = (_Float16)(c * xn[i]);
        *(v8h*)(xo + e * DIMC + kb) = h;
    }
}

// ------------------------------------------------------------- WMMA GEMM
// C[M,N] = A[M,K] @ B[K,N] (+bias)(+resid)(gelu), f16 in, f32 accum.
// Block: 256 threads = 8 wave32. Block tile: 128(M) x 64(N), BK=32.
// Wave macro-tile: 32x32 = 2x2 WMMA tiles (4 accs, frag reuse 2x, 2 LDS
// loads per WMMA). Waves arranged 4(M) x 2(N).
// EPI: 0 = plain -> f16 ; 1 = +bias +resid -> f32 ; 2 = +bias, gelu -> f16
template <int EPI>
__global__ __launch_bounds__(256)
void gemm_wmma_kernel(const _Float16* __restrict__ A, int lda,
                      const _Float16* __restrict__ Bw, int ldb,
                      const float* __restrict__ bias,
                      const float* __restrict__ resid,
                      float* __restrict__ outF, _Float16* __restrict__ outH,
                      int M, int N, int K) {
    __shared__ _Float16 sA[128 * 40];   // 128 rows x 32 k, stride 40 (pad)
    __shared__ _Float16 sB[64 * 40];    // 64 n x 32 k (transposed), stride 40

    const int tid  = threadIdx.x;
    const int lane = tid & 31;
    const int wid  = tid >> 5;
    const int m0   = blockIdx.y * 128;
    const int n0   = blockIdx.x * 64;

    // staging coords: A = 128x32 (2 x v8h per thread), B = 32x64 (v8h/thread)
    const int arow  = tid >> 1;
    const int akoff = (tid & 1) * 16;
    const int bk    = tid >> 3;          // 0..31 (k)
    const int bn    = (tid & 7) * 8;     // 0..56 (n, 8 halves)

    // fragment coords (ISA wave32 WMMA layouts)
    const int mbase = (wid >> 1) * 32;   // wave M position (4 slots)
    const int nbase = (wid & 1) * 32;    // wave N position (2 slots)
    const int frow = lane & 15;
    const int fkbA = (lane >> 4) * 8;    // A: halves [0:8)=k+0..7, [8:16)=k+16..23
    const int fn   = lane & 15;
    const int fkbB = (lane >> 4) * 8;    // B: halves [0:8)=k+0..7, [8:16)=k+16..23

    v8f acc00 = {}, acc01 = {}, acc10 = {}, acc11 = {};

    for (int k0 = 0; k0 < K; k0 += 32) {
        const _Float16* ag = A + (size_t)(m0 + arow) * lda + k0 + akoff;
        v8h a0 = *(const v8h*)ag;
        v8h a1 = *(const v8h*)(ag + 8);
        const _Float16* bg = Bw + (size_t)(k0 + bk) * ldb + n0 + bn;
        v8h bv = *(const v8h*)bg;

        __syncthreads();   // previous iteration's LDS reads done
        *(v8h*)(&sA[arow * 40 + akoff])     = a0;
        *(v8h*)(&sA[arow * 40 + akoff + 8]) = a1;
#pragma unroll
        for (int j = 0; j < 8; j++) sB[(bn + j) * 40 + bk] = bv[j];
        __syncthreads();

        if (k0 + 32 < K) {   // warm caches for next tiles (global_prefetch_b8)
            __builtin_prefetch(ag + 32, 0, 1);
            __builtin_prefetch(bg + (size_t)32 * ldb, 0, 1);
        }

        // A fragments for the wave's two M tiles
        const _Float16* apm0 = &sA[(mbase + frow) * 40 + fkbA];
        const _Float16* apm1 = &sA[(mbase + 16 + frow) * 40 + fkbA];
        v8h a0l = *(const v8h*)apm0;
        v8h a0h = *(const v8h*)(apm0 + 16);
        v8h a1l = *(const v8h*)apm1;
        v8h a1h = *(const v8h*)(apm1 + 16);
        v16h afr0 = __builtin_shufflevector(a0l, a0h, 0, 1, 2, 3, 4, 5, 6, 7,
                                            8, 9, 10, 11, 12, 13, 14, 15);
        v16h afr1 = __builtin_shufflevector(a1l, a1h, 0, 1, 2, 3, 4, 5, 6, 7,
                                            8, 9, 10, 11, 12, 13, 14, 15);
        // B fragments for the wave's two N tiles
        const _Float16* bpn0 = &sB[(nbase + fn) * 40 + fkbB];
        const _Float16* bpn1 = &sB[(nbase + 16 + fn) * 40 + fkbB];
        v8h b0l = *(const v8h*)bpn0;
        v8h b0h = *(const v8h*)(bpn0 + 16);
        v8h b1l = *(const v8h*)bpn1;
        v8h b1h = *(const v8h*)(bpn1 + 16);
        v16h bfr0 = __builtin_shufflevector(b0l, b0h, 0, 1, 2, 3, 4, 5, 6, 7,
                                            8, 9, 10, 11, 12, 13, 14, 15);
        v16h bfr1 = __builtin_shufflevector(b1l, b1h, 0, 1, 2, 3, 4, 5, 6, 7,
                                            8, 9, 10, 11, 12, 13, 14, 15);

        acc00 = __builtin_amdgcn_wmma_f32_16x16x32_f16(
            false, afr0, false, bfr0, (short)0, acc00, false, false);
        acc01 = __builtin_amdgcn_wmma_f32_16x16x32_f16(
            false, afr0, false, bfr1, (short)0, acc01, false, false);
        acc10 = __builtin_amdgcn_wmma_f32_16x16x32_f16(
            false, afr1, false, bfr0, (short)0, acc10, false, false);
        acc11 = __builtin_amdgcn_wmma_f32_16x16x32_f16(
            false, afr1, false, bfr1, (short)0, acc11, false, false);
    }

#pragma unroll
    for (int jm = 0; jm < 2; jm++) {
#pragma unroll
        for (int jn = 0; jn < 2; jn++) {
            v8f acc = jm ? (jn ? acc11 : acc10) : (jn ? acc01 : acc00);
            const int gn = n0 + nbase + jn * 16 + fn;
            const float bvl = (EPI > 0) ? bias[gn] : 0.f;
#pragma unroll
            for (int i = 0; i < 8; i++) {
                const int gm = m0 + mbase + jm * 16 + i + ((lane >> 4) << 3);
                float c = acc[i] + bvl;
                if (EPI == 1) c += resid[(size_t)gm * N + gn];
                if (EPI == 2) c = 0.5f * c * (1.f + erff(c * 0.70710678118f));
                if (EPI == 1) outF[(size_t)gm * N + gn] = c;
                else          outH[(size_t)gm * N + gn] = (_Float16)c;
            }
        }
    }
}

// ------------------------------------------------- per-token 4x4 attention
// y[B,768] f16 -> s = q+k+v (aliased buffers in torch ref) -> softmax(s s^T/8) s
// writes T[tok, d*4 + h] f16 (the transpose(1,2).reshape layout)
__global__ __launch_bounds__(256)
void attn_kernel(const _Float16* __restrict__ y, _Float16* __restrict__ T) {
    const int lane = threadIdx.x & 31;
    const int tok  = blockIdx.x * 8 + (threadIdx.x >> 5);
    const _Float16* yr = y + (size_t)tok * 768;

    float s[NHEAD][2];
#pragma unroll
    for (int h = 0; h < NHEAD; h++)
#pragma unroll
        for (int j = 0; j < 2; j++) {
            const int d = lane * 2 + j;
            s[h][j] = (float)yr[h * DHEAD + d] + (float)yr[256 + h * DHEAD + d] +
                      (float)yr[512 + h * DHEAD + d];
        }

    float dot[NHEAD][NHEAD];
#pragma unroll
    for (int h = 0; h < NHEAD; h++)
#pragma unroll
        for (int g = 0; g < NHEAD; g++) {
            if (g < h) continue;
            float p = s[h][0] * s[g][0] + s[h][1] * s[g][1];
            p = wave_sum(p);
            dot[h][g] = p;
            dot[g][h] = p;
        }

    const float scale = 0.125f;   // 64^-0.5
    float attn[NHEAD][NHEAD];
#pragma unroll
    for (int h = 0; h < NHEAD; h++) {
        float mx = dot[h][0];
#pragma unroll
        for (int g = 1; g < NHEAD; g++) mx = fmaxf(mx, dot[h][g]);
        float ss = 0.f;
#pragma unroll
        for (int g = 0; g < NHEAD; g++) {
            attn[h][g] = __expf((dot[h][g] - mx) * scale);
            ss += attn[h][g];
        }
        const float inv = 1.f / ss;
#pragma unroll
        for (int g = 0; g < NHEAD; g++) attn[h][g] *= inv;
    }

#pragma unroll
    for (int j = 0; j < 2; j++) {
        const int d = lane * 2 + j;
        v4h o;
#pragma unroll
        for (int h = 0; h < NHEAD; h++) {
            float ov = 0.f;
#pragma unroll
            for (int g = 0; g < NHEAD; g++) ov += attn[h][g] * s[g][j];
            o[h] = (_Float16)ov;
        }
        *(v4h*)(T + (size_t)tok * DIMC + d * 4) = o;
    }
}

// ------------------------------------------------------------- layernorm2
__global__ __launch_bounds__(256)
void ln2_kernel(const float* __restrict__ x1, const float* __restrict__ g2,
                const float* __restrict__ b2, _Float16* __restrict__ out) {
    const int lane = threadIdx.x & 31;
    const int tok  = blockIdx.x * 8 + (threadIdx.x >> 5);
    const int kb   = lane * 8;
    const float* xr = x1 + (size_t)tok * DIMC;
    float v[8];
#pragma unroll
    for (int i = 0; i < 8; i++) v[i] = xr[kb + i];
    float s = 0.f;
#pragma unroll
    for (int i = 0; i < 8; i++) s += v[i];
    const float mean = wave_sum(s) * (1.f / 256.f);
    float q = 0.f;
#pragma unroll
    for (int i = 0; i < 8; i++) { float d = v[i] - mean; q += d * d; }
    const float rstd = rsqrtf(wave_sum(q) * (1.f / 256.f) + 1e-5f);
    v8h h;
#pragma unroll
    for (int i = 0; i < 8; i++)
        h[i] = (_Float16)((v[i] - mean) * rstd * g2[kb + i] + b2[kb + i]);
    *(v8h*)(out + (size_t)tok * DIMC + kb) = h;
}

// ---------------------------------------------------------------- launcher
extern "C" void kernel_launch(void* const* d_in, const int* in_sizes, int n_in,
                              void* d_out, int out_size, void* d_ws, size_t ws_size,
                              hipStream_t stream) {
    const float* x      = (const float*)d_in[0];
    const float* gate_w = (const float*)d_in[1];
    const float* gate_b = (const float*)d_in[2];
    const float* qkv_w  = (const float*)d_in[3];
    const float* proj_w = (const float*)d_in[4];
    const float* proj_b = (const float*)d_in[5];
    const float* n1g    = (const float*)d_in[6];
    const float* n1b    = (const float*)d_in[7];
    const float* n2g    = (const float*)d_in[8];
    const float* n2b    = (const float*)d_in[9];
    const float* fc1_w  = (const float*)d_in[10];
    const float* fc1_b  = (const float*)d_in[11];
    const float* fc2_w  = (const float*)d_in[12];
    const float* fc2_b  = (const float*)d_in[13];
    float* out = (float*)d_out;

    char* ws = (char*)d_ws;
    size_t off = 0;
    auto alloc = [&](size_t bytes) {
        void* p = ws + off;
        off = (off + bytes + 255) & ~(size_t)255;
        return p;
    };
    _Float16* xs = (_Float16*)alloc((size_t)TOK * 1024 * 2);  // 128 MB
    char*     yb = (char*)    alloc((size_t)TOK * 768 * 2);   //  96 MB (aliased)
    _Float16* T  = (_Float16*)alloc((size_t)TOK * 256 * 2);   //  32 MB
    _Float16* wq = (_Float16*)alloc((size_t)1024 * 768 * 2);
    _Float16* wp = (_Float16*)alloc((size_t)256 * 256 * 2);
    _Float16* w1 = (_Float16*)alloc((size_t)256 * 1024 * 2);
    _Float16* w2 = (_Float16*)alloc((size_t)1024 * 256 * 2);

    _Float16* y   = (_Float16*)yb;                         // [B,768] f16
    float*    x1  = (float*)yb;                            // reuse after attn
    _Float16* h16 = (_Float16*)(yb + (size_t)TOK * 256 * 4);
    _Float16* gel = xs;                                    // reuse after qkv gemm

    // weight conversions
    cvt_f32_f16_kernel<<<(1024 * 768 + 255) / 256, 256, 0, stream>>>(qkv_w, wq, 1024 * 768);
    cvt_f32_f16_kernel<<<(256 * 256 + 255) / 256, 256, 0, stream>>>(proj_w, wp, 256 * 256);
    cvt_f32_f16_kernel<<<(256 * 1024 + 255) / 256, 256, 0, stream>>>(fc1_w, w1, 256 * 1024);
    cvt_f32_f16_kernel<<<(1024 * 256 + 255) / 256, 256, 0, stream>>>(fc2_w, w2, 1024 * 256);

    // LN1 + gate + top2 -> gate-scaled per-expert rows
    ln_gate_kernel<<<TOK / 8, 256, 0, stream>>>(x, n1g, n1b, gate_w, gate_b, xs);

    // qkv: [B,1024] @ [1024,768] -> y f16
    gemm_wmma_kernel<0><<<dim3(768 / 64, TOK / 128), 256, 0, stream>>>(
        xs, 1024, wq, 768, nullptr, nullptr, nullptr, y, TOK, 768, 1024);

    // per-token 4x4 attention -> transposed proj input
    attn_kernel<<<TOK / 8, 256, 0, stream>>>(y, T);

    // proj: [B,256] @ [256,256] + proj_b + x -> x1 f32
    gemm_wmma_kernel<1><<<dim3(256 / 64, TOK / 128), 256, 0, stream>>>(
        T, 256, wp, 256, proj_b, x, x1, nullptr, TOK, 256, 256);

    // LN2 -> h16
    ln2_kernel<<<TOK / 8, 256, 0, stream>>>(x1, n2g, n2b, h16);

    // fc1 + gelu: [B,256] @ [256,1024] -> gel f16
    gemm_wmma_kernel<2><<<dim3(1024 / 64, TOK / 128), 256, 0, stream>>>(
        h16, 256, w1, 1024, fc1_b, nullptr, nullptr, gel, TOK, 1024, 256);

    // fc2 + residual: [B,1024] @ [1024,256] + fc2_b + x1 -> out f32
    gemm_wmma_kernel<1><<<dim3(256 / 64, TOK / 128), 256, 0, stream>>>(
        gel, 1024, w2, 256, fc2_b, x1, out, nullptr, TOK, 256, 1024);
}
